// GaussianEmbeddingDP_34093450396016
// MI455X (gfx1250) — matrix operand order repaired and checked by reading, not compiled, not run on hardware
//
#include <hip/hip_runtime.h>
#include <math.h>

typedef __attribute__((ext_vector_type(16))) _Float16 v16h;
typedef __attribute__((ext_vector_type(8)))  float    v8f;

#define N_ROWS    16384
#define N_COLS    1024
#define HALF_ROWS 8192
#define HALF_N    8388608u   // (16384*1024)/2 : JAX threefry pairs e with e+HALF_N

// ---------------- JAX-compatible threefry2x32 with key = (0, 42) -------------
__device__ __forceinline__ void threefry2x32_key42(unsigned int x0, unsigned int x1,
                                                   unsigned int& o0, unsigned int& o1) {
  const unsigned int k0 = 0u, k1 = 42u;
  const unsigned int k2 = k0 ^ k1 ^ 0x1BD11BDAu;
  unsigned int v0 = x0 + k0;
  unsigned int v1 = x1 + k1;
#define TF_ROUND(r) do { v0 += v1; v1 = (v1 << (r)) | (v1 >> (32 - (r))); v1 ^= v0; } while (0)
  TF_ROUND(13); TF_ROUND(15); TF_ROUND(26); TF_ROUND(6);
  v0 += k1; v1 += k2 + 1u;
  TF_ROUND(17); TF_ROUND(29); TF_ROUND(16); TF_ROUND(24);
  v0 += k2; v1 += k0 + 2u;
  TF_ROUND(13); TF_ROUND(15); TF_ROUND(26); TF_ROUND(6);
  v0 += k0; v1 += k1 + 3u;
  TF_ROUND(17); TF_ROUND(29); TF_ROUND(16); TF_ROUND(24);
  v0 += k1; v1 += k2 + 4u;
  TF_ROUND(13); TF_ROUND(15); TF_ROUND(26); TF_ROUND(6);
  v0 += k2; v1 += k0 + 5u;
#undef TF_ROUND
  o0 = v0; o1 = v1;
}

// Giles single-precision erfinv (matches XLA's ErfInv lowering)
__device__ __forceinline__ float erfinv_f32(float x) {
  float w = -__logf((1.0f - x) * (1.0f + x));
  float p;
  if (w < 5.0f) {
    w -= 2.5f;
    p =              2.81022636e-08f;
    p = fmaf(p, w,   3.43273939e-07f);
    p = fmaf(p, w,  -3.5233877e-06f);
    p = fmaf(p, w,  -4.39150654e-06f);
    p = fmaf(p, w,   0.00021858087f);
    p = fmaf(p, w,  -0.00125372503f);
    p = fmaf(p, w,  -0.00417768164f);
    p = fmaf(p, w,   0.246640727f);
    p = fmaf(p, w,   1.50140941f);
  } else {
    w = sqrtf(w) - 3.0f;
    p =             -0.000200214257f;
    p = fmaf(p, w,   0.000100950558f);
    p = fmaf(p, w,   0.00134934322f);
    p = fmaf(p, w,  -0.00367342844f);
    p = fmaf(p, w,   0.00573950773f);
    p = fmaf(p, w,  -0.0076224613f);
    p = fmaf(p, w,   0.00943887047f);
    p = fmaf(p, w,   1.00167406f);
    p = fmaf(p, w,   2.83297682f);
  }
  return p * x;
}

// bits -> uniform(-1,1) (JAX mantissa trick + minval clamp) -> N(0,1)
__device__ __forceinline__ float bits_to_normal(unsigned int bits) {
  float f = __uint_as_float((bits >> 9) | 0x3f800000u) - 1.0f;   // [0,1)
  const float lo = -0.99999994f;                                  // nextafter(-1,0)
  float u = fmaf(f, 1.99999988f, lo);                             // f*(1-lo)+lo
  u = fmaxf(u, lo);
  return 1.41421356f * erfinv_f32(u);
}

// ----------------------------- fused kernel ---------------------------------
// Each wave owns 16 rows: frag rows 0..7 -> global rows [lowBase, lowBase+8)
//                          frag rows 8..15 -> mirror rows +8192 (threefry pairing)
// Pass A: Gram-diagonal norms via V_WMMA_F32_16X16X32_F16 (f32 accumulate).
// Pass B: re-stream rows (L2-resident), scale, add threefry noise (2 outputs/call).
__global__ __launch_bounds__(256) void dp_clip_noise_kernel(const float* __restrict__ x,
                                                            float* __restrict__ out) {
  const int lane    = threadIdx.x & 31;
  const int wave    = threadIdx.x >> 5;
  const int gwave   = blockIdx.x * 8 + wave;   // 0..1023
  const int lowBase = gwave * 8;               // 8 low-half rows per wave

  // ---- Pass A: norms via WMMA gram diagonal ----
  const int m    = lane & 15;                                           // frag row
  const int rowM = (m < 8) ? (lowBase + m) : (HALF_ROWS + lowBase + (m - 8));
  const float* rowPtr = x + (size_t)rowM * N_COLS;

  const int kbA = (lane < 16) ? 0 : 16;   // A-frag K base per lane half (contiguous k = 16h + s)
  const int kbB = (lane < 16) ? 0 : 16;   // B-frag K base per lane half (contiguous k = 16h + s, same as A)

  v8f acc = {0.f, 0.f, 0.f, 0.f, 0.f, 0.f, 0.f, 0.f};

  for (int k0 = 0; k0 < N_COLS; k0 += 32) {
    __builtin_prefetch(rowPtr + k0 + 128, 0, 0);   // global_prefetch ahead of stream
    const float4 a0 = *(const float4*)(rowPtr + k0 + kbA);
    const float4 a1 = *(const float4*)(rowPtr + k0 + kbA + 4);
    const float4 a2 = *(const float4*)(rowPtr + k0 + kbA + 8);
    const float4 a3 = *(const float4*)(rowPtr + k0 + kbA + 12);
    const float4 b0 = *(const float4*)(rowPtr + k0 + kbB);
    const float4 b1 = *(const float4*)(rowPtr + k0 + kbB + 4);
    const float4 b2 = *(const float4*)(rowPtr + k0 + kbB + 8);
    const float4 b3 = *(const float4*)(rowPtr + k0 + kbB + 12);

    v16h af, bf;
    af[0]  = (_Float16)a0.x; af[1]  = (_Float16)a0.y; af[2]  = (_Float16)a0.z; af[3]  = (_Float16)a0.w;
    af[4]  = (_Float16)a1.x; af[5]  = (_Float16)a1.y; af[6]  = (_Float16)a1.z; af[7]  = (_Float16)a1.w;
    af[8]  = (_Float16)a2.x; af[9]  = (_Float16)a2.y; af[10] = (_Float16)a2.z; af[11] = (_Float16)a2.w;
    af[12] = (_Float16)a3.x; af[13] = (_Float16)a3.y; af[14] = (_Float16)a3.z; af[15] = (_Float16)a3.w;
    bf[0]  = (_Float16)b0.x; bf[1]  = (_Float16)b0.y; bf[2]  = (_Float16)b0.z; bf[3]  = (_Float16)b0.w;
    bf[4]  = (_Float16)b1.x; bf[5]  = (_Float16)b1.y; bf[6]  = (_Float16)b1.z; bf[7]  = (_Float16)b1.w;
    bf[8]  = (_Float16)b2.x; bf[9]  = (_Float16)b2.y; bf[10] = (_Float16)b2.z; bf[11] = (_Float16)b2.w;
    bf[12] = (_Float16)b3.x; bf[13] = (_Float16)b3.y; bf[14] = (_Float16)b3.z; bf[15] = (_Float16)b3.w;

    // D = A * A^T + C : diagonal accumulates per-row sum of squares
    acc = __builtin_amdgcn_wmma_f32_16x16x32_f16(false, af, false, bf,
                                                 (short)0, acc, false, false);
  }

  // Extract Gram diagonal (C/D layout: m<8 -> VGPR m @ lane m; m>=8 -> VGPR m-8 @ lane m+16)
  float scale[16];
#pragma unroll
  for (int r = 0; r < 8; ++r) {
    float ss = __shfl(acc[r], r, 32);
    scale[r] = 1.0f / fmaxf(sqrtf(ss), 1.0f);
  }
#pragma unroll
  for (int r = 8; r < 16; ++r) {
    float ss = __shfl(acc[r - 8], r + 16, 32);
    scale[r] = 1.0f / fmaxf(sqrtf(ss), 1.0f);
  }

  // ---- Pass B: scale + noise, rows re-read through L2 (input fits in 192MB L2) ----
#pragma unroll
  for (int j = 0; j < 8; ++j) {
    const int rLow  = lowBase + j;
    const int rHigh = rLow + HALF_ROWS;
    const float sLo = scale[j];
    const float sHi = scale[j + 8];
    const float* pLo = x   + (size_t)rLow  * N_COLS;
    const float* pHi = x   + (size_t)rHigh * N_COLS;
    float*       oLo = out + (size_t)rLow  * N_COLS;
    float*       oHi = out + (size_t)rHigh * N_COLS;

    for (int c = lane * 4; c < N_COLS; c += 128) {
      const float4 vlo = *(const float4*)(pLo + c);
      const float4 vhi = *(const float4*)(pHi + c);
      const unsigned int e0 = (unsigned int)rLow * N_COLS + (unsigned int)c;

      float4 rl, rh;
      unsigned int o0, o1;
      // one threefry block yields noise for row r (out0) and row r+8192 (out1)
      threefry2x32_key42(e0 + 0u, e0 + 0u + HALF_N, o0, o1);
      rl.x = fmaf(1.1f, bits_to_normal(o0), vlo.x * sLo);
      rh.x = fmaf(1.1f, bits_to_normal(o1), vhi.x * sHi);
      threefry2x32_key42(e0 + 1u, e0 + 1u + HALF_N, o0, o1);
      rl.y = fmaf(1.1f, bits_to_normal(o0), vlo.y * sLo);
      rh.y = fmaf(1.1f, bits_to_normal(o1), vhi.y * sHi);
      threefry2x32_key42(e0 + 2u, e0 + 2u + HALF_N, o0, o1);
      rl.z = fmaf(1.1f, bits_to_normal(o0), vlo.z * sLo);
      rh.z = fmaf(1.1f, bits_to_normal(o1), vhi.z * sHi);
      threefry2x32_key42(e0 + 3u, e0 + 3u + HALF_N, o0, o1);
      rl.w = fmaf(1.1f, bits_to_normal(o0), vlo.w * sLo);
      rh.w = fmaf(1.1f, bits_to_normal(o1), vhi.w * sHi);

      *(float4*)(oLo + c) = rl;
      *(float4*)(oHi + c) = rh;
    }
  }
}

extern "C" void kernel_launch(void* const* d_in, const int* in_sizes, int n_in,
                              void* d_out, int out_size, void* d_ws, size_t ws_size,
                              hipStream_t stream) {
  (void)in_sizes; (void)n_in; (void)out_size; (void)d_ws; (void)ws_size;
  const float* x  = (const float*)d_in[0];
  float*       out = (float*)d_out;
  // 128 blocks x 8 waves x 16 rows/wave = 16384 rows
  hipLaunchKernelGGL(dp_clip_noise_kernel, dim3(128), dim3(256), 0, stream, x, out);
}
